// RelativeTransformerEncoderLayer_16252156248180
// MI455X (gfx1250) — matrix operand order repaired and checked by reading, not compiled, not run on hardware
//
#include <hip/hip_runtime.h>
#include <stdint.h>

// ---------------- problem constants ----------------
#define Q_LEN  1024
#define BATCH  8
#define DMODEL 1024
#define NHEAD  16
#define DHEAD  64
#define FFDIM  4096
#define NTOK   (Q_LEN * BATCH)     // 8192
#define BH_CHUNK 16                // (b,h) pairs per attention chunk
#define ATT_SCALE 0.125f           // 1/sqrt(64)

typedef __attribute__((ext_vector_type(16))) __bf16 v16bf;
typedef __attribute__((ext_vector_type(8)))  float  v8f;
typedef __attribute__((ext_vector_type(4)))  int    v4i;

// ---- CDNA5 async global->LDS copy (probe-guarded, sync fallback) ----
#if defined(__has_builtin)
#if __has_builtin(__builtin_amdgcn_global_load_async_to_lds_b128)
#define HAVE_ASYNC_LDS 1
#endif
#endif
#ifndef HAVE_ASYNC_LDS
#define HAVE_ASYNC_LDS 0
#endif

#if HAVE_ASYNC_LDS
__device__ __forceinline__ void async_b128(const void* g, void* l) {
  __builtin_amdgcn_global_load_async_to_lds_b128(
      (__attribute__((address_space(1))) v4i*)g,
      (__attribute__((address_space(3))) v4i*)l, 0, 0);
}
#if __has_builtin(__builtin_amdgcn_s_wait_asynccnt)
#define WAIT_ASYNC_IF() __builtin_amdgcn_s_wait_asynccnt(0)
#else
#define WAIT_ASYNC_IF() asm volatile("s_wait_asynccnt 0" ::: "memory")
#endif
#else
#define WAIT_ASYNC_IF() do {} while (0)
#endif

union BF16Frag {
  v16bf v;
  unsigned short u[16];
  uint4 q[2];
};

__device__ __forceinline__ unsigned short f2bf(float f) {
  union { float f; unsigned int u; } c; c.f = f;
  unsigned int u = c.u;
  return (unsigned short)((u + 0x7FFFu + ((u >> 16) & 1u)) >> 16);  // RNE
}

__device__ __forceinline__ v8f zero8() {
  v8f z = {0.f, 0.f, 0.f, 0.f, 0.f, 0.f, 0.f, 0.f};
  return z;
}

__device__ __forceinline__ v8f wmma_bf16(v16bf a, v16bf b, v8f c) {
  return __builtin_amdgcn_wmma_f32_16x16x32_bf16(false, a, false, b, (short)0, c,
                                                 false, false);
}

// ---------------- elementwise kernels ----------------
__global__ __launch_bounds__(256) void f32_to_bf16_kernel(
    const float* __restrict__ in, unsigned short* __restrict__ out, size_t n) {
  size_t i = (size_t)blockIdx.x * 256 + threadIdx.x;
  if (i < n) out[i] = f2bf(in[i]);
}

__global__ __launch_bounds__(256) void ln_bf16_kernel(
    const float* __restrict__ x, const float* __restrict__ g,
    const float* __restrict__ bb, unsigned short* __restrict__ out) {
  __shared__ float red[256];
  const int row = blockIdx.x;
  const int tid = threadIdx.x;
  const float* xr = x + (size_t)row * DMODEL;
  float v[4];
  float s = 0.f;
#pragma unroll
  for (int i = 0; i < 4; ++i) { v[i] = xr[tid + i * 256]; s += v[i]; }
  red[tid] = s; __syncthreads();
  for (int st = 128; st > 0; st >>= 1) {
    if (tid < st) red[tid] += red[tid + st];
    __syncthreads();
  }
  const float mu = red[0] * (1.f / DMODEL);
  __syncthreads();
  s = 0.f;
#pragma unroll
  for (int i = 0; i < 4; ++i) { float d = v[i] - mu; s += d * d; }
  red[tid] = s; __syncthreads();
  for (int st = 128; st > 0; st >>= 1) {
    if (tid < st) red[tid] += red[tid + st];
    __syncthreads();
  }
  const float rstd = rsqrtf(red[0] * (1.f / DMODEL) + 1e-5f);
#pragma unroll
  for (int i = 0; i < 4; ++i) {
    int c = tid + i * 256;
    out[(size_t)row * DMODEL + c] = f2bf((v[i] - mu) * rstd * g[c] + bb[c]);
  }
}

// ---------------- generic WMMA GEMM (double-buffered, async staging) -------
// C[M,N] = A[M,K](bf16,row-major,lda) * B[K,N](bf16,row-major,ldb), fp32 acc.
#define BM 128
#define BN 128
#define BK 32

__global__ __launch_bounds__(256) void gemm_bf16_kernel(
    const unsigned short* __restrict__ A, int lda,
    const unsigned short* __restrict__ B, int ldb, int K,
    const float* __restrict__ bias, const float* __restrict__ residual,
    float* __restrict__ outF, unsigned short* __restrict__ outB, int ldc,
    int relu, int qkvMode,
    const float* __restrict__ rwb, const float* __restrict__ rrb,
    unsigned short* __restrict__ wqc, unsigned short* __restrict__ wqr,
    unsigned short* __restrict__ wk, unsigned short* __restrict__ wv) {
  __shared__ __align__(16) unsigned short As[2][BM][BK + 8];
  __shared__ __align__(16) unsigned short BsT[2][BN][BK + 8];
  const int tid = threadIdx.x;
  const int lane = tid & 31;
  const int wid = tid >> 5;
  const int wm = wid >> 2;   // 0..1
  const int wn = wid & 3;    // 0..3
  const int lm = lane & 15;
  const int lh = lane >> 4;  // 0/1
  const int m0 = blockIdx.y * BM;
  const int n0 = blockIdx.x * BN;

  // per-thread staging coordinates (fully unrolled, 2 chunks each)
  const int ar0 = tid >> 2;            // 0..63
  const int ar1 = ar0 + 64;
  const int ac0 = (tid & 3) * 8;
  const int bk0 = tid >> 4;            // 0..15
  const int bk1 = bk0 + 16;
  const int bn0 = (tid & 15) * 8;

  const unsigned short* aP0 = A + (size_t)(m0 + ar0) * lda + ac0;
  const unsigned short* aP1 = A + (size_t)(m0 + ar1) * lda + ac0;
  const unsigned short* bP0 = B + (size_t)bk0 * ldb + n0 + bn0;
  const unsigned short* bP1 = B + (size_t)bk1 * ldb + n0 + bn0;
  const size_t bstep = (size_t)BK * ldb;

  uint4 breg0, breg1;
#if !HAVE_ASYNC_LDS
  uint4 areg0, areg1;
#endif

#if HAVE_ASYNC_LDS
  auto stageA = [&](int buf) {
    async_b128(aP0, &As[buf][ar0][ac0]);
    async_b128(aP1, &As[buf][ar1][ac0]);
    aP0 += BK; aP1 += BK;
  };
  auto commitA = [&](int) {};
#else
  auto stageA = [&](int) {
    areg0 = *(const uint4*)aP0;
    areg1 = *(const uint4*)aP1;
    aP0 += BK; aP1 += BK;
  };
  auto commitA = [&](int buf) {
    *(uint4*)&As[buf][ar0][ac0] = areg0;
    *(uint4*)&As[buf][ar1][ac0] = areg1;
  };
#endif
  auto loadB = [&]() {
    breg0 = *(const uint4*)bP0;
    breg1 = *(const uint4*)bP1;
    bP0 += bstep; bP1 += bstep;
  };
  auto storeB = [&](int buf) {  // transpose-commit: BsT[n][k]
    const unsigned short* p0 = (const unsigned short*)&breg0;
    const unsigned short* p1 = (const unsigned short*)&breg1;
#pragma unroll
    for (int e = 0; e < 8; ++e) {
      BsT[buf][bn0 + e][bk0] = p0[e];
      BsT[buf][bn0 + e][bk1] = p1[e];
    }
  };

  v8f acc[4][2];
#pragma unroll
  for (int i = 0; i < 4; ++i)
#pragma unroll
    for (int j = 0; j < 2; ++j) acc[i][j] = zero8();

  // pipeline prologue: tile 0
  stageA(0);
  loadB();
  commitA(0);
  storeB(0);
  WAIT_ASYNC_IF();
  __syncthreads();

  const int S = K / BK;
  for (int step = 0; step < S; ++step) {
    const int cur = step & 1;
    const int nxt = cur ^ 1;
    const bool more = (step + 1 < S);
    if (more) { stageA(nxt); loadB(); }

    BF16Frag af[4], bf[2];
#pragma unroll
    for (int ms = 0; ms < 4; ++ms) {
      int row = wm * 64 + ms * 16 + lm;
      int ko = lh * 8;  // lanes<16: K{0..7,16..23}; lanes>=16: K{8..15,24..31}
      af[ms].q[0] = *(const uint4*)&As[cur][row][ko];
      af[ms].q[1] = *(const uint4*)&As[cur][row][16 + ko];
    }
#pragma unroll
    for (int ns = 0; ns < 2; ++ns) {
      int col = wn * 32 + ns * 16 + lm;
      int ko = lh * 8;  // lanes<16: K{0..7,16..23}; lanes>=16: K{8..15,24..31}
      bf[ns].q[0] = *(const uint4*)&BsT[cur][col][ko];
      bf[ns].q[1] = *(const uint4*)&BsT[cur][col][ko + 16];
    }
#pragma unroll
    for (int ms = 0; ms < 4; ++ms)
#pragma unroll
      for (int ns = 0; ns < 2; ++ns)
        acc[ms][ns] = wmma_bf16(af[ms].v, bf[ns].v, acc[ms][ns]);

    if (more) {
      commitA(nxt);
      storeB(nxt);
      WAIT_ASYNC_IF();
      __syncthreads();
    }
  }

#pragma unroll
  for (int ms = 0; ms < 4; ++ms)
#pragma unroll
    for (int ns = 0; ns < 2; ++ns)
#pragma unroll
      for (int v = 0; v < 8; ++v) {
        int row = m0 + wm * 64 + ms * 16 + v + 8 * lh;
        int col = n0 + wn * 32 + ns * 16 + lm;
        float val = acc[ms][ns][v];
        if (bias) val += bias[col];
        if (qkvMode) {
          if (col < DMODEL) {
            wqc[(size_t)row * DMODEL + col] = f2bf(val + rwb[col]);
            wqr[(size_t)row * DMODEL + col] = f2bf(val + rrb[col]);
          } else if (col < 2 * DMODEL) {
            wk[(size_t)row * DMODEL + (col - DMODEL)] = f2bf(val);
          } else {
            wv[(size_t)row * DMODEL + (col - 2 * DMODEL)] = f2bf(val);
          }
        } else {
          if (relu) val = fmaxf(val, 0.f);
          if (residual) val += residual[(size_t)row * ldc + col];
          if (outF) outF[(size_t)row * ldc + col] = val;
          else outB[(size_t)row * ldc + col] = f2bf(val);
        }
      }
}

// ---------------- batched BDraw GEMM (async both operands) ----------------
// bd[z][i][r] = sum_d wqr[i,b,h,d] * rk[r,h,d]   (bh = bh0 + z)
__global__ __launch_bounds__(256) void bd_gemm_kernel(
    const unsigned short* __restrict__ wqr,
    const unsigned short* __restrict__ rkbf, float* __restrict__ bd, int bh0) {
  __shared__ __align__(16) unsigned short As[2][BM][BK + 8];
  __shared__ __align__(16) unsigned short BsT[2][BN][BK + 8];
  const int tid = threadIdx.x;
  const int lane = tid & 31;
  const int wid = tid >> 5;
  const int wm = wid >> 2, wn = wid & 3;
  const int lm = lane & 15, lh = lane >> 4;
  const int m0 = blockIdx.y * BM;
  const int n0 = blockIdx.x * BN;
  const int bh = bh0 + blockIdx.z;
  const int h = bh & (NHEAD - 1);
  const size_t lda = (size_t)BATCH * DMODEL;  // 8192

  const int ar0 = tid >> 2, ar1 = (tid >> 2) + 64;
  const int ac0 = (tid & 3) * 8;
  const int rn0 = tid >> 2, rn1 = (tid >> 2) + 64;
  const int rc0 = (tid & 3) * 8;

  const unsigned short* aP0 =
      wqr + (size_t)bh * DHEAD + (size_t)(m0 + ar0) * lda + ac0;
  const unsigned short* aP1 =
      wqr + (size_t)bh * DHEAD + (size_t)(m0 + ar1) * lda + ac0;
  // rk rows are k-contiguous: transposed LDS layout is a straight b128 copy
  const unsigned short* rP0 =
      rkbf + ((size_t)(n0 + rn0) * NHEAD + h) * DHEAD + rc0;
  const unsigned short* rP1 =
      rkbf + ((size_t)(n0 + rn1) * NHEAD + h) * DHEAD + rc0;

#if !HAVE_ASYNC_LDS
  uint4 areg0, areg1, rreg0, rreg1;
#endif

#if HAVE_ASYNC_LDS
  auto stageA = [&](int buf) {
    async_b128(aP0, &As[buf][ar0][ac0]);
    async_b128(aP1, &As[buf][ar1][ac0]);
    aP0 += BK; aP1 += BK;
  };
  auto stageB = [&](int buf) {
    async_b128(rP0, &BsT[buf][rn0][rc0]);
    async_b128(rP1, &BsT[buf][rn1][rc0]);
    rP0 += BK; rP1 += BK;
  };
  auto commitAB = [&](int) {};
#else
  auto stageA = [&](int) {
    areg0 = *(const uint4*)aP0; areg1 = *(const uint4*)aP1;
    aP0 += BK; aP1 += BK;
  };
  auto stageB = [&](int) {
    rreg0 = *(const uint4*)rP0; rreg1 = *(const uint4*)rP1;
    rP0 += BK; rP1 += BK;
  };
  auto commitAB = [&](int buf) {
    *(uint4*)&As[buf][ar0][ac0] = areg0;
    *(uint4*)&As[buf][ar1][ac0] = areg1;
    *(uint4*)&BsT[buf][rn0][rc0] = rreg0;
    *(uint4*)&BsT[buf][rn1][rc0] = rreg1;
  };
#endif

  v8f acc[4][2];
#pragma unroll
  for (int i = 0; i < 4; ++i)
#pragma unroll
    for (int j = 0; j < 2; ++j) acc[i][j] = zero8();

  stageA(0); stageB(0);
  commitAB(0);
  WAIT_ASYNC_IF();
  __syncthreads();

  const int S = DHEAD / BK;  // 2
#pragma unroll
  for (int step = 0; step < S; ++step) {
    const int cur = step & 1;
    const int nxt = cur ^ 1;
    const bool more = (step + 1 < S);
    if (more) { stageA(nxt); stageB(nxt); }

    BF16Frag af[4], bf[2];
#pragma unroll
    for (int ms = 0; ms < 4; ++ms) {
      int row = wm * 64 + ms * 16 + lm;
      int ko = lh * 8;
      af[ms].q[0] = *(const uint4*)&As[cur][row][ko];
      af[ms].q[1] = *(const uint4*)&As[cur][row][16 + ko];
    }
#pragma unroll
    for (int ns = 0; ns < 2; ++ns) {
      int col = wn * 32 + ns * 16 + lm;
      int ko = lh * 8;
      bf[ns].q[0] = *(const uint4*)&BsT[cur][col][ko];
      bf[ns].q[1] = *(const uint4*)&BsT[cur][col][ko + 16];
    }
#pragma unroll
    for (int ms = 0; ms < 4; ++ms)
#pragma unroll
      for (int ns = 0; ns < 2; ++ns)
        acc[ms][ns] = wmma_bf16(af[ms].v, bf[ns].v, acc[ms][ns]);

    if (more) {
      commitAB(nxt);
      WAIT_ASYNC_IF();
      __syncthreads();
    }
  }

  float* bz = bd + (size_t)blockIdx.z * Q_LEN * Q_LEN;
#pragma unroll
  for (int ms = 0; ms < 4; ++ms)
#pragma unroll
    for (int ns = 0; ns < 2; ++ns)
#pragma unroll
      for (int v = 0; v < 8; ++v) {
        int row = m0 + wm * 64 + ms * 16 + v + 8 * lh;
        int col = n0 + wn * 32 + ns * 16 + lm;
        bz[(size_t)row * Q_LEN + col] = acc[ms][ns][v];
      }
}

// ---------------- flash attention with Transformer-XL rel-shift ----------------
// One wave = one 16-row q-tile of one (b,h). Block = 4 waves.
__global__ __launch_bounds__(128) void attn_kernel(
    const unsigned short* __restrict__ wqc, const unsigned short* __restrict__ wk,
    const unsigned short* __restrict__ wv, const float* __restrict__ bd,
    const unsigned char* __restrict__ mask, unsigned short* __restrict__ attn,
    int bh0) {
  __shared__ __align__(16) unsigned short pshare[4][16][16];
  const int lane = threadIdx.x & 31;
  const int w = threadIdx.x >> 5;
  const int lm = lane & 15;
  const int lh = lane >> 4;
  const int i0 = (blockIdx.x * 4 + w) * 16;
  const int bh = bh0 + blockIdx.y;
  const int b = bh >> 4;          // NHEAD = 16
  const int h = bh & (NHEAD - 1);
  const float* bdp = bd + (size_t)blockIdx.y * Q_LEN * Q_LEN;

  // Qc A-fragments (16x64 split as 2 x K=32)
  BF16Frag qa[2];
  {
    const unsigned short* qrow =
        wqc + (((size_t)(i0 + lm) * BATCH + b) * NHEAD + h) * DHEAD;
#pragma unroll
    for (int f = 0; f < 2; ++f) {
      int d0 = f * 32 + lh * 8;
      qa[f].q[0] = *(const uint4*)(qrow + d0);
      qa[f].q[1] = *(const uint4*)(qrow + d0 + 16);
    }
  }

  v8f o[4];
#pragma unroll
  for (int nt = 0; nt < 4; ++nt) o[nt] = zero8();
  float m[8], lsum[8];
#pragma unroll
  for (int v = 0; v < 8; ++v) { m[v] = -INFINITY; lsum[v] = 0.f; }

  for (int j0 = 0; j0 < Q_LEN; j0 += 16) {
    // K^T B-fragments: column n = K-row j0+lm, contiguous along DHEAD
    BF16Frag kb[2];
    {
      const unsigned short* krow =
          wk + (((size_t)(j0 + lm) * BATCH + b) * NHEAD + h) * DHEAD;
#pragma unroll
      for (int f = 0; f < 2; ++f) {
        int d0 = f * 32 + lh * 8;
        kb[f].q[0] = *(const uint4*)(krow + d0);
        kb[f].q[1] = *(const uint4*)(krow + d0 + 16);
      }
    }
    v8f s = zero8();
    s = wmma_bf16(qa[0].v, kb[0].v, s);
    s = wmma_bf16(qa[1].v, kb[1].v, s);

    // add rel-shifted BD, scale, mask
    float sv[8];
#pragma unroll
    for (int v = 0; v < 8; ++v) {
      int i = i0 + v + 8 * lh;
      int j = j0 + lm;
      float bdv;
      if (j <= i)          bdv = bdp[(size_t)i * Q_LEN + (Q_LEN - 1 + j - i)];
      else if (j == i + 1) bdv = 0.f;
      else                 bdv = bdp[(size_t)(i + 1) * Q_LEN + (j - i - 2)];
      float val = (s[v] + bdv) * ATT_SCALE;
      if (mask[(size_t)i * Q_LEN + j]) val = -1e30f;
      sv[v] = val;
    }

    // online softmax (row stats live in the 16 lanes of each half-wave)
    float fac[8];
#pragma unroll
    for (int v = 0; v < 8; ++v) {
      float rmax = sv[v];
#pragma unroll
      for (int off = 8; off >= 1; off >>= 1)
        rmax = fmaxf(rmax, __shfl_xor(rmax, off, 32));
      float mn = fmaxf(m[v], rmax);
      float p = __expf(sv[v] - mn);
      float rsum = p;
#pragma unroll
      for (int off = 8; off >= 1; off >>= 1) rsum += __shfl_xor(rsum, off, 32);
      fac[v] = __expf(m[v] - mn);
      m[v] = mn;
      lsum[v] = lsum[v] * fac[v] + rsum;
      sv[v] = p;
    }
#pragma unroll
    for (int nt = 0; nt < 4; ++nt)
#pragma unroll
      for (int v = 0; v < 8; ++v) o[nt][v] *= fac[v];

    // P (C-layout) -> LDS -> A-fragment layout, K zero-padded to 32
#pragma unroll
    for (int v = 0; v < 8; ++v) pshare[w][v + 8 * lh][lm] = f2bf(sv[v]);
    asm volatile("s_wait_dscnt 0" ::: "memory");
    BF16Frag pa;
    pa.q[0] = *(const uint4*)&pshare[w][lm][lh * 8];
    {
      uint4 z4; z4.x = z4.y = z4.z = z4.w = 0u;
      pa.q[1] = z4;
    }
    asm volatile("" ::: "memory");  // keep load before next-iter store

    // O += P @ V ; only elements 0..7 of each lane (K 8*lh+0..7) carry real V data
#pragma unroll
    for (int nt = 0; nt < 4; ++nt) {
      BF16Frag vb;
      {
        int n = nt * 16 + lm;
#pragma unroll
        for (int k = 0; k < 8; ++k)
          vb.u[k] = wv[(((size_t)(j0 + 8 * lh + k) * BATCH + b) * NHEAD + h) * DHEAD + n];
#pragma unroll
        for (int k = 8; k < 16; ++k) vb.u[k] = 0;
      }
      o[nt] = wmma_bf16(pa.v, vb.v, o[nt]);
    }
  }

  // normalize and store attn (bf16, [Q,B,H,DH])
#pragma unroll
  for (int v = 0; v < 8; ++v) {
    float inv = 1.f / lsum[v];
    int i = i0 + v + 8 * lh;
    unsigned short* arow = attn + (((size_t)i * BATCH + b) * NHEAD + h) * DHEAD;
#pragma unroll
    for (int nt = 0; nt < 4; ++nt) arow[nt * 16 + lm] = f2bf(o[nt][v] * inv);
  }
}

// ---------------- host orchestration ----------------
extern "C" void kernel_launch(void* const* d_in, const int* in_sizes, int n_in,
                              void* d_out, int out_size, void* d_ws,
                              size_t ws_size, hipStream_t stream) {
  (void)in_sizes; (void)n_in; (void)out_size; (void)ws_size;
  const float* x       = (const float*)d_in[0];
  const float* pos     = (const float*)d_in[1];
  const unsigned char* mask = (const unsigned char*)d_in[2];
  const float* ln1_g   = (const float*)d_in[3];
  const float* ln1_b   = (const float*)d_in[4];
  const float* qkv_w   = (const float*)d_in[5];
  const float* qkv_b   = (const float*)d_in[6];
  const float* r_w     = (const float*)d_in[7];
  const float* r_w_bias = (const float*)d_in[8];
  const float* r_r_bias = (const float*)d_in[9];
  const float* o_w     = (const float*)d_in[10];
  const float* ln2_g   = (const float*)d_in[11];
  const float* ln2_b   = (const float*)d_in[12];
  const float* f_w1    = (const float*)d_in[13];
  const float* f_b1    = (const float*)d_in[14];
  const float* f_w2    = (const float*)d_in[15];
  const float* f_b2    = (const float*)d_in[16];
  float* out = (float*)d_out;
  char* ws = (char*)d_ws;

  // workspace layout (bytes)
  size_t off = 0;
  auto take = [&](size_t bytes) {
    size_t o = off;
    off += (bytes + 255) & ~(size_t)255;
    return o;
  };
  unsigned short* wqkvB = (unsigned short*)(ws + take((size_t)DMODEL * 3 * DMODEL * 2));
  unsigned short* wrwB  = (unsigned short*)(ws + take((size_t)DMODEL * DMODEL * 2));
  unsigned short* wowB  = (unsigned short*)(ws + take((size_t)DMODEL * DMODEL * 2));
  unsigned short* wf1B  = (unsigned short*)(ws + take((size_t)DMODEL * FFDIM * 2));
  unsigned short* wf2B  = (unsigned short*)(ws + take((size_t)FFDIM * DMODEL * 2));
  unsigned short* posB  = (unsigned short*)(ws + take((size_t)Q_LEN * DMODEL * 2));
  unsigned short* lnB   = (unsigned short*)(ws + take((size_t)NTOK * DMODEL * 2));
  unsigned short* wqcB  = (unsigned short*)(ws + take((size_t)NTOK * DMODEL * 2));
  unsigned short* wqrB  = (unsigned short*)(ws + take((size_t)NTOK * DMODEL * 2));
  unsigned short* wkB   = (unsigned short*)(ws + take((size_t)NTOK * DMODEL * 2));
  unsigned short* wvB   = (unsigned short*)(ws + take((size_t)NTOK * DMODEL * 2));
  unsigned short* rkB   = (unsigned short*)(ws + take((size_t)Q_LEN * DMODEL * 2));
  unsigned short* attnB = (unsigned short*)(ws + take((size_t)NTOK * DMODEL * 2));
  // reused region: bd chunk (64MB) then ffn hidden (64MB); x1 after it (32MB)
  size_t r1 = take((size_t)BH_CHUNK * Q_LEN * Q_LEN * 4);  // == NTOK*FFDIM*2
  float*          bdF = (float*)(ws + r1);
  unsigned short* h1B = (unsigned short*)(ws + r1);
  float*          x1F = (float*)(ws + take((size_t)NTOK * DMODEL * 4));

  // 1) fp32 -> bf16 weight/pos conversion
  auto cvt = [&](const float* src, unsigned short* dst, size_t n) {
    f32_to_bf16_kernel<<<(unsigned)((n + 255) / 256), 256, 0, stream>>>(src, dst, n);
  };
  cvt(qkv_w, wqkvB, (size_t)DMODEL * 3 * DMODEL);
  cvt(r_w,   wrwB,  (size_t)DMODEL * DMODEL);
  cvt(o_w,   wowB,  (size_t)DMODEL * DMODEL);
  cvt(f_w1,  wf1B,  (size_t)DMODEL * FFDIM);
  cvt(f_w2,  wf2B,  (size_t)FFDIM * DMODEL);
  cvt(pos,   posB,  (size_t)Q_LEN * DMODEL);

  // 2) LN1
  ln_bf16_kernel<<<NTOK, 256, 0, stream>>>(x, ln1_g, ln1_b, lnB);

  // 3) QKV GEMM with fused split + head biases
  gemm_bf16_kernel<<<dim3(3 * DMODEL / BN, NTOK / BM), 256, 0, stream>>>(
      lnB, DMODEL, wqkvB, 3 * DMODEL, DMODEL, qkv_b, nullptr, nullptr, nullptr,
      0, 0, 1, r_w_bias, r_r_bias, wqcB, wqrB, wkB, wvB);

  // 4) rk = pos_emb @ r_w  -> bf16 [Q,H,DH]
  gemm_bf16_kernel<<<dim3(DMODEL / BN, Q_LEN / BM), 256, 0, stream>>>(
      posB, DMODEL, wrwB, DMODEL, DMODEL, nullptr, nullptr, nullptr, rkB,
      DMODEL, 0, 0, nullptr, nullptr, nullptr, nullptr, nullptr, nullptr);

  // 5) attention, chunked over (b,h)
  for (int c = 0; c < (BATCH * NHEAD) / BH_CHUNK; ++c) {
    int bh0 = c * BH_CHUNK;
    bd_gemm_kernel<<<dim3(Q_LEN / BN, Q_LEN / BM, BH_CHUNK), 256, 0, stream>>>(
        wqrB, rkB, bdF, bh0);
    attn_kernel<<<dim3(Q_LEN / 64, BH_CHUNK), 128, 0, stream>>>(
        wqcB, wkB, wvB, bdF, mask, attnB, bh0);
  }

  // 6) output projection + residual -> x1 (fp32)
  gemm_bf16_kernel<<<dim3(DMODEL / BN, NTOK / BM), 256, 0, stream>>>(
      attnB, DMODEL, wowB, DMODEL, DMODEL, nullptr, x, x1F, nullptr, DMODEL, 0,
      0, nullptr, nullptr, nullptr, nullptr, nullptr, nullptr);

  // 7) LN2
  ln_bf16_kernel<<<NTOK, 256, 0, stream>>>(x1F, ln2_g, ln2_b, lnB);

  // 8) FFN1: relu(ln2 @ W1 + b1) -> bf16 h1
  gemm_bf16_kernel<<<dim3(FFDIM / BN, NTOK / BM), 256, 0, stream>>>(
      lnB, DMODEL, wf1B, FFDIM, DMODEL, f_b1, nullptr, nullptr, h1B, FFDIM, 1,
      0, nullptr, nullptr, nullptr, nullptr, nullptr, nullptr);

  // 9) FFN2: h1 @ W2 + b2 + x1 -> out (fp32)
  gemm_bf16_kernel<<<dim3(DMODEL / BN, NTOK / BM), 256, 0, stream>>>(
      h1B, FFDIM, wf2B, DMODEL, FFDIM, f_b2, x1F, out, nullptr, DMODEL, 0, 0,
      nullptr, nullptr, nullptr, nullptr, nullptr, nullptr);
}
